// MatchingNetwork_27539330302226
// MI455X (gfx1250) — matrix operand order repaired and checked by reading, not compiled, not run on hardware
//
#include <hip/hip_runtime.h>
#include <math.h>

typedef __attribute__((ext_vector_type(16))) _Float16 v16h;
typedef __attribute__((ext_vector_type(8)))  float    v8f;
typedef __attribute__((ext_vector_type(2)))  float    v2f;

static __device__ __forceinline__ float sigmf(float x) { return 1.0f / (1.0f + __expf(-x)); }

// ---------------------------------------------------------------------------
// Pack an f32 matrix W[K][N] into CDNA5 f16 B-fragment layout:
//   out[kc][nt][lane(32)][16 halves], K padded to Kc*32 with zeros.
// Fragment element e=2i+j of lane L (g=L>>4, n=nt*16+(L&15)) holds
//   W[kc*32 + g*16 + 2i + j][n]   (B rows striped: lanes 0-15 K lo, 16-31 K hi)
// ---------------------------------------------------------------------------
__global__ void pack_bfrag_f16(const float* __restrict__ W, _Float16* __restrict__ out,
                               int Kreal, int N, int Kc) {
  int NT = N >> 4;
  int total = Kc * NT * 32;
  int idx = blockIdx.x * blockDim.x + threadIdx.x;
  if (idx >= total) return;
  int lane = idx & 31;
  int nt = (idx >> 5) % NT;
  int kc = idx / (32 * NT);
  int g = lane >> 4, n = nt * 16 + (lane & 15);
  _Float16* o = out + (size_t)idx * 16;
#pragma unroll
  for (int i = 0; i < 8; i++)
#pragma unroll
    for (int j = 0; j < 2; j++) {
      int K = kc * 32 + g * 16 + 2 * i + j;
      float v = (K < Kreal) ? W[(size_t)K * N + n] : 0.0f;
      o[2 * i + j] = (_Float16)v;
    }
}

// Pack LSTM recurrent weights Wr[32][128] into f32 B-fragment layout for
// v_wmma_f32_16x16x4_f32: out[ks(8)][nt(8)][lane(32)][2 floats],
// reg i of lane L holds Wr[ks*4 + (L>>4)*2 + i][nt*16 + (L&15)].
__global__ void pack_rec_f32(const float* __restrict__ Wr, float* __restrict__ out) {
  int idx = blockIdx.x * blockDim.x + threadIdx.x;  // 8*8*32 = 2048
  if (idx >= 2048) return;
  int lane = idx & 31;
  int nt = (idx >> 5) & 7;
  int ks = idx >> 8;
  int g = lane >> 4, n = nt * 16 + (lane & 15);
#pragma unroll
  for (int i = 0; i < 2; i++) {
    int K = ks * 4 + g * 2 + i;
    out[idx * 2 + i] = Wr[K * 128 + n];
  }
}

// ---------------------------------------------------------------------------
// Fused conv3x3(SAME) + bias + BN + ReLU + maxpool2x2(VALID), implicit GEMM.
// One wave per tile: 16 output pixels (2 conv rows x 8 conv cols) x 64 chans.
// Branchless im2col: addresses clamped into the image, loads unconditional,
// zeros selected via cndmask. CIN templated so k-decomposition is mul/shift.
// Each lane stages its 16 contiguous halves as one v16h LDS vector store.
// ---------------------------------------------------------------------------
template <int CIN>
__global__ void __launch_bounds__(32)
conv_bn_relu_pool(const float* __restrict__ in, float* __restrict__ out,
                  const _Float16* __restrict__ bfrag,
                  int Hin, int Win, int Hp, int Wp,
                  int Kreal, int Kc, int tilesX,
                  const float* __restrict__ cb, const float* __restrict__ bg,
                  const float* __restrict__ bb, const float* __restrict__ bm,
                  const float* __restrict__ bv) {
  constexpr int C3 = 3 * CIN;
  __shared__ _Float16 At[16 * 32];
  int lane = threadIdx.x;
  int ty = blockIdx.x / tilesX, cx = blockIdx.x % tilesX;
  int img = blockIdx.y;
  const float* src = in + (size_t)img * Hin * Win * CIN;
  v8f acc[4] = {};
  int m = lane & 15;
  int khalf = lane >> 4;
  int row = 2 * ty + (m >> 3);
  int col = 8 * cx + (m & 7);
  for (int kc = 0; kc < Kc; kc++) {
    if (kc + 1 < Kc) __builtin_prefetch(bfrag + ((size_t)(kc + 1) * 4 * 32) * 16, 0, 1);
    // Branchless im2col fill: each lane builds 16 contiguous f16 values.
    v16h av;
#pragma unroll
    for (int t = 0; t < 16; t++) {
      int kg = kc * 32 + khalf * 16 + t;
      int kh = kg / C3;              // constant divisor -> mul/shift
      int r2 = kg - kh * C3;
      int kw = r2 / CIN;
      int ci = r2 - kw * CIN;
      int iy = row + kh - 1, ix = col + kw - 1;
      bool ok = (kg < Kreal) & (iy >= 0) & (iy < Hin) & (ix >= 0) & (ix < Win);
      int cy = min(max(iy, 0), Hin - 1);
      int cx2 = min(max(ix, 0), Win - 1);
      float v = src[((size_t)cy * Win + cx2) * CIN + ci];  // always valid addr
      av[t] = (_Float16)(ok ? v : 0.0f);
    }
    *(v16h*)(At + m * 32 + khalf * 16) = av;
    __syncthreads();
    // Gather A fragment in contiguous k order (k = 16*khalf + i), matching the B packing
    v16h a;
    {
      int klo = khalf * 16;
#pragma unroll
      for (int i = 0; i < 16; i++) {
        int kk = klo + i;
        a[i] = At[m * 32 + kk];
      }
    }
    const _Float16* bf = bfrag + ((size_t)kc * 4 * 32) * 16;
#pragma unroll
    for (int nt = 0; nt < 4; nt++) {
      v16h b = *(const v16h*)(bf + ((size_t)nt * 32 + lane) * 16);
      acc[nt] = __builtin_amdgcn_wmma_f32_16x16x32_f16(false, a, false, b, (short)0,
                                                       acc[nt], false, false);
    }
    __syncthreads();
  }
  // Epilogue: bias + BN + ReLU + 2x2 max pool, write pooled activations.
  int g2 = lane >> 4, n0 = lane & 15;
#pragma unroll
  for (int nt = 0; nt < 4; nt++) {
    int f = nt * 16 + n0;
    float s = bg[f] * rsqrtf(bv[f] + 1e-3f);
    float off = bb[f] - bm[f] * s;
    float bias = cb[f];
    float pm[4];
#pragma unroll
    for (int j = 0; j < 4; j++) {
      float v0 = fmaxf((acc[nt][2 * j] + bias) * s + off, 0.0f);
      float v1 = fmaxf((acc[nt][2 * j + 1] + bias) * s + off, 0.0f);
      pm[j] = fmaxf(v0, v1);
    }
#pragma unroll
    for (int j = 0; j < 4; j++) pm[j] = fmaxf(pm[j], __shfl_xor(pm[j], 16));
    if (g2 == 0) {
#pragma unroll
      for (int j = 0; j < 4; j++) {
        int px = 4 * cx + j;
        if (px < Wp)
          out[(((size_t)img * Hp + ty) * Wp + px) * 64 + f] = pm[j];
      }
    }
  }
}

// ---------------------------------------------------------------------------
// Row-major f32 A [M][K] x pre-packed f16 B fragments -> f32 C [M][ldc].
// One wave computes a 16 x 64 tile. Requires M % 16 == 0 and K % 32 == 0
// (true at all call sites), so the A fill is 4x float4 loads, no guards.
// ---------------------------------------------------------------------------
__global__ void __launch_bounds__(32)
gemm_rows_f16(const float* __restrict__ A, int M, int K, int Kc,
              const _Float16* __restrict__ bfrag, int NTtotal,
              float* __restrict__ C, int ldc) {
  __shared__ _Float16 At[16 * 32];
  int lane = threadIdx.x;
  int m0 = blockIdx.x * 16;
  int nb = blockIdx.y;
  v8f acc[4] = {};
  int m = lane & 15, khalf = lane >> 4;
  const float* arow = A + (size_t)(m0 + m) * K + khalf * 16;
  for (int kc = 0; kc < Kc; kc++) {
    if (kc + 1 < Kc)
      __builtin_prefetch(bfrag + ((size_t)(kc + 1) * NTtotal + nb * 4) * 32 * 16, 0, 1);
    v16h av;
#pragma unroll
    for (int t = 0; t < 16; t += 4) {
      float4 f = *(const float4*)(arow + kc * 32 + t);
      av[t] = (_Float16)f.x; av[t + 1] = (_Float16)f.y;
      av[t + 2] = (_Float16)f.z; av[t + 3] = (_Float16)f.w;
    }
    *(v16h*)(At + m * 32 + khalf * 16) = av;
    __syncthreads();
    v16h a;
    int klo = khalf * 16;
#pragma unroll
    for (int i = 0; i < 16; i++) a[i] = At[m * 32 + klo + i];
    const _Float16* bf = bfrag + ((size_t)kc * NTtotal + nb * 4) * 32 * 16;
#pragma unroll
    for (int nt = 0; nt < 4; nt++) {
      v16h b = *(const v16h*)(bf + ((size_t)nt * 32 + lane) * 16);
      acc[nt] = __builtin_amdgcn_wmma_f32_16x16x32_f16(false, a, false, b, (short)0,
                                                       acc[nt], false, false);
    }
    __syncthreads();
  }
  int g2 = lane >> 4, n0 = lane & 15;
#pragma unroll
  for (int nt = 0; nt < 4; nt++)
#pragma unroll
    for (int r = 0; r < 8; r++) {
      int rowi = m0 + r + 8 * g2;
      int coli = nb * 64 + nt * 16 + n0;
      if (rowi < M) C[(size_t)rowi * ldc + coli] = acc[nt][r];
    }
}

// Build LSTM GEMM input matrices in (t,b)/(q,b) row order from embeddings.
__global__ void gather_seq_rows(const float* __restrict__ emb,
                                float* __restrict__ Asup, float* __restrict__ Aq) {
  int idx = blockIdx.x * blockDim.x + threadIdx.x;
  if (idx >= 640 * 1600) return;
  int row = idx / 1600, k = idx - row * 1600;
  if (row < 400) {
    int s = row >> 4, b = row & 15;
    Asup[idx] = emb[((size_t)(b * 25 + s)) * 1600 + k];
  } else {
    int r = row - 400;
    int q = r >> 4, b = r & 15;
    Aq[(size_t)r * 1600 + k] = emb[((size_t)(400 + b * 15 + q)) * 1600 + k];
  }
}

// ---------------------------------------------------------------------------
// LSTM scan: one wave per (query, direction). Each step: z(16x128) =
// xk[t] + h @ Wr + bias via v_wmma_f32_16x16x4_f32 (8 N-tiles x 8 K-steps),
// then fused gates. h/c kept in LDS/registers; hidden states written out.
// ---------------------------------------------------------------------------
__global__ void __launch_bounds__(32)
lstm_scan(const float* __restrict__ xk_sup_f, const float* __restrict__ xk_sup_b,
          const float* __restrict__ xkq_f, const float* __restrict__ xkq_b,
          const float* __restrict__ bias_f, const float* __restrict__ bias_b,
          const float* __restrict__ recfrag, float* __restrict__ hs) {
  __shared__ float wr[4096];
  __shared__ float hl[16 * 32];
  __shared__ float zl[16 * 128];
  int lane = threadIdx.x;
  int q = blockIdx.x, dir = blockIdx.y;
  const float* xk_sup = dir ? xk_sup_b : xk_sup_f;
  const float* xkq    = dir ? xkq_b : xkq_f;
  const float* bias   = dir ? bias_b : bias_f;
  const float* rf = recfrag + dir * 4096;
  float* hsout = hs + ((size_t)(dir * 15 + q)) * 26 * 512;
  for (int i = lane; i < 4096; i += 32) wr[i] = rf[i];
  for (int i = lane; i < 512; i += 32) hl[i] = 0.0f;
  float c[16];
#pragma unroll
  for (int i = 0; i < 16; i++) c[i] = 0.0f;
  __syncthreads();
  int m = lane & 15, g = lane >> 4, n0 = lane & 15;
  for (int step = 0; step < 26; step++) {
    int ot = dir ? (25 - step) : step;
    const float* xrow = (ot < 25) ? (xk_sup + (size_t)ot * 2048) : (xkq + (size_t)q * 2048);
#pragma unroll 2
    for (int nt = 0; nt < 8; nt++) {
      v8f acc = {};
#pragma unroll
      for (int ks = 0; ks < 8; ks++) {
        v2f a, b;
        int k0 = ks * 4 + g * 2;
        a.x = hl[m * 32 + k0];
        a.y = hl[m * 32 + k0 + 1];
        const float* bp = wr + ((ks * 8 + nt) * 32 + lane) * 2;
        b.x = bp[0]; b.y = bp[1];
        acc = __builtin_amdgcn_wmma_f32_16x16x4_f32(false, a, false, b, (short)0,
                                                    acc, false, false);
      }
#pragma unroll
      for (int r = 0; r < 8; r++) {
        int mm = r + 8 * g;
        int coli = nt * 16 + n0;
        zl[mm * 128 + coli] = acc[r] + xrow[mm * 128 + coli] + bias[coli];
      }
    }
    __syncthreads();
    int j0 = g * 16;
#pragma unroll
    for (int jj = 0; jj < 16; jj++) {
      int j = j0 + jj;
      float zi = zl[m * 128 + j];
      float zf = zl[m * 128 + 32 + j];
      float zg = zl[m * 128 + 64 + j];
      float zo = zl[m * 128 + 96 + j];
      float cn = sigmf(zf) * c[jj] + sigmf(zi) * tanhf(zg);
      float hn = sigmf(zo) * tanhf(cn);
      c[jj] = cn;
      hl[m * 32 + j] = hn;
      hsout[((size_t)ot * 16 + m) * 32 + j] = hn;
    }
    __syncthreads();
  }
}

// Attention: one wave per (q,b); lane s handles support item s.
__global__ void __launch_bounds__(32)
attention_kernel(const float* __restrict__ hs, const int* __restrict__ ysup,
                 float* __restrict__ preds) {
  __shared__ float pr[5];
  int lane = threadIdx.x;
  int q = blockIdx.x / 16, b = blockIdx.x % 16;
  const float* hsF = hs + ((size_t)q) * 26 * 512;
  const float* hsB = hs + ((size_t)(15 + q)) * 26 * 512;
  float logit = -INFINITY;
  if (lane < 25) {
    float dot = 0.0f, ss = 0.0f;
    const float* qF = hsF + ((size_t)25 * 16 + b) * 32;
    const float* qB = hsB + ((size_t)25 * 16 + b) * 32;
    const float* sF = hsF + ((size_t)lane * 16 + b) * 32;
    const float* sB = hsB + ((size_t)lane * 16 + b) * 32;
    for (int j = 0; j < 32; j++) { float v = sF[j]; dot += v * qF[j]; ss += v * v; }
    for (int j = 0; j < 32; j++) { float v = sB[j]; dot += v * qB[j]; ss += v * v; }
    logit = dot * rsqrtf(fmaxf(ss, 1e-10f));
  }
  float mx = logit;
  for (int o = 16; o > 0; o >>= 1) mx = fmaxf(mx, __shfl_xor(mx, o));
  float e = (lane < 25) ? __expf(logit - mx) : 0.0f;
  float sum = e;
  for (int o = 16; o > 0; o >>= 1) sum += __shfl_xor(sum, o);
  float sim = e / sum;
  if (lane < 5) pr[lane] = 0.0f;
  __syncthreads();
  if (lane < 25) atomicAdd(&pr[ysup[b * 25 + lane]], sim);
  __syncthreads();
  if (lane < 5) preds[((size_t)q * 16 + b) * 5 + lane] = pr[lane];
}

// CE-per-batch-mean over queries + accuracy.
__global__ void __launch_bounds__(32)
finalize_kernel(const float* __restrict__ preds, const int* __restrict__ yq,
                float* __restrict__ out) {
  __shared__ float ceb[16];
  __shared__ int corr;
  int lane = threadIdx.x;
  if (lane < 16) ceb[lane] = 0.0f;
  if (lane == 0) corr = 0;
  __syncthreads();
  for (int idx = lane; idx < 240; idx += 32) {
    int q = idx / 16, b = idx & 15;
    float p[5]; float s = 0.0f;
    for (int w = 0; w < 5; w++) { p[w] = preds[idx * 5 + w]; s += p[w]; }
    int label = yq[b * 15 + q];
    float pn = p[label] / s;
    pn = fminf(fmaxf(pn, 1e-7f), 1.0f - 1e-7f);
    atomicAdd(&ceb[b], -__logf(pn));
    int am = 0; float bmax = p[0];
    for (int w = 1; w < 5; w++) if (p[w] > bmax) { bmax = p[w]; am = w; }
    if (am == label) atomicAdd(&corr, 1);
  }
  __syncthreads();
  if (lane < 16) out[lane] = ceb[lane] / 15.0f;
  __syncthreads();
  if (lane == 0) out[16] = (float)corr / 240.0f;
}

// ---------------------------------------------------------------------------
extern "C" void kernel_launch(void* const* d_in, const int* in_sizes, int n_in,
                              void* d_out, int out_size, void* d_ws, size_t ws_size,
                              hipStream_t stream) {
  (void)in_sizes; (void)n_in; (void)out_size; (void)ws_size;
  const float* x_sup = (const float*)d_in[0];
  const float* x_qry = (const float*)d_in[1];
  const int*   y_sup = (const int*)d_in[2];
  const int*   y_qry = (const int*)d_in[3];
  const float* w1 = (const float*)d_in[4];
  const float* w2 = (const float*)d_in[5];
  const float* w3 = (const float*)d_in[6];
  const float* w4 = (const float*)d_in[7];
  const float* cb = (const float*)d_in[8];
  const float* bg = (const float*)d_in[9];
  const float* bbt = (const float*)d_in[10];
  const float* bmn = (const float*)d_in[11];
  const float* bvr = (const float*)d_in[12];
  const float* wkf = (const float*)d_in[13];
  const float* wrf = (const float*)d_in[14];
  const float* biasf = (const float*)d_in[15];
  const float* wkb = (const float*)d_in[16];
  const float* wrb = (const float*)d_in[17];
  const float* biasb = (const float*)d_in[18];

  char* base = (char*)d_ws;
  size_t off = 0;
  auto alloc = [&](size_t bytes) -> char* {
    char* p = base + off;
    off = (off + bytes + 255) & ~(size_t)255;
    return p;
  };
  float* A0  = (float*)alloc(640ull * 84 * 84 * 3 * 4);
  float* P1  = (float*)alloc(640ull * 42 * 42 * 64 * 4);
  float* P2  = (float*)alloc(640ull * 21 * 21 * 64 * 4);
  float* P3  = (float*)alloc(640ull * 10 * 10 * 64 * 4);
  float* EMB = (float*)alloc(640ull * 1600 * 4);
  _Float16* WF1 = (_Float16*)alloc(1ull * 4 * 32 * 16 * 2);
  _Float16* WF2 = (_Float16*)alloc(18ull * 4 * 32 * 16 * 2);
  _Float16* WF3 = (_Float16*)alloc(18ull * 4 * 32 * 16 * 2);
  _Float16* WF4 = (_Float16*)alloc(18ull * 4 * 32 * 16 * 2);
  _Float16* WKF = (_Float16*)alloc(50ull * 8 * 32 * 16 * 2);
  _Float16* WKB = (_Float16*)alloc(50ull * 8 * 32 * 16 * 2);
  float* RECF = (float*)alloc(2ull * 4096 * 4);
  float* ASUP = (float*)alloc(400ull * 1600 * 4);
  float* AQ   = (float*)alloc(240ull * 1600 * 4);
  float* XKSF = (float*)alloc(400ull * 128 * 4);
  float* XKSB = (float*)alloc(400ull * 128 * 4);
  float* XKQF = (float*)alloc(240ull * 128 * 4);
  float* XKQB = (float*)alloc(240ull * 128 * 4);
  float* HS   = (float*)alloc(2ull * 15 * 26 * 512 * 4);
  float* PREDS = (float*)alloc(240ull * 5 * 4);

  // Concatenate support+query images into one activation buffer (D2D async).
  hipMemcpyAsync(A0, x_sup, 400ull * 21168 * 4, hipMemcpyDeviceToDevice, stream);
  hipMemcpyAsync(A0 + 400ull * 21168, x_qry, 240ull * 21168 * 4,
                 hipMemcpyDeviceToDevice, stream);

  auto packL = [&](const float* W, _Float16* o, int Kreal, int N, int Kc) {
    int total = Kc * (N / 16) * 32;
    pack_bfrag_f16<<<(total + 255) / 256, 256, 0, stream>>>(W, o, Kreal, N, Kc);
  };
  packL(w1, WF1, 27, 64, 1);
  packL(w2, WF2, 576, 64, 18);
  packL(w3, WF3, 576, 64, 18);
  packL(w4, WF4, 576, 64, 18);
  packL(wkf, WKF, 1600, 128, 50);
  packL(wkb, WKB, 1600, 128, 50);
  pack_rec_f32<<<8, 256, 0, stream>>>(wrf, RECF);
  pack_rec_f32<<<8, 256, 0, stream>>>(wrb, RECF + 4096);

  // Encoder: 4 fused conv+BN+ReLU+pool layers.
  conv_bn_relu_pool<3><<<dim3(42 * 11, 640), 32, 0, stream>>>(
      A0, P1, WF1, 84, 84, 42, 42, 27, 1, 11, cb, bg, bbt, bmn, bvr);
  conv_bn_relu_pool<64><<<dim3(21 * 6, 640), 32, 0, stream>>>(
      P1, P2, WF2, 42, 42, 21, 21, 576, 18, 6,
      cb + 64, bg + 64, bbt + 64, bmn + 64, bvr + 64);
  conv_bn_relu_pool<64><<<dim3(10 * 3, 640), 32, 0, stream>>>(
      P2, P3, WF3, 21, 21, 10, 10, 576, 18, 3,
      cb + 128, bg + 128, bbt + 128, bmn + 128, bvr + 128);
  conv_bn_relu_pool<64><<<dim3(5 * 2, 640), 32, 0, stream>>>(
      P3, EMB, WF4, 10, 10, 5, 5, 576, 18, 2,
      cb + 192, bg + 192, bbt + 192, bmn + 192, bvr + 192);

  // LSTM input projections (shared across queries, hoisted out of the scan).
  gather_seq_rows<<<(640 * 1600 + 255) / 256, 256, 0, stream>>>(EMB, ASUP, AQ);
  gemm_rows_f16<<<dim3(25, 2), 32, 0, stream>>>(ASUP, 400, 1600, 50, WKF, 8, XKSF, 128);
  gemm_rows_f16<<<dim3(25, 2), 32, 0, stream>>>(ASUP, 400, 1600, 50, WKB, 8, XKSB, 128);
  gemm_rows_f16<<<dim3(15, 2), 32, 0, stream>>>(AQ, 240, 1600, 50, WKF, 8, XKQF, 128);
  gemm_rows_f16<<<dim3(15, 2), 32, 0, stream>>>(AQ, 240, 1600, 50, WKB, 8, XKQB, 128);

  // Bidirectional recurrence (f32 WMMA), then attention + readout.
  lstm_scan<<<dim3(15, 2), 32, 0, stream>>>(XKSF, XKSB, XKQF, XKQB,
                                            biasf, biasb, RECF, HS);
  attention_kernel<<<240, 32, 0, stream>>>(HS, y_sup, PREDS);
  finalize_kernel<<<1, 32, 0, stream>>>(PREDS, y_qry, (float*)d_out);
}
